// AstrocyteAssociativeMemory_42700564857153
// MI455X (gfx1250) — compile-verified, matrix operand order checked by reading, not run on hardware
//
#include <hip/hip_runtime.h>
#include <hip/hip_bf16.h>
#include <math.h>

#define DD    2048
#define MM    50000
#define BATCH 16384
#define KTOP  5
#define LPITCH 40   // LDS row pitch in halves (80B) — keeps 16B alignment, spreads banks

typedef __attribute__((ext_vector_type(16))) _Float16 v16h;
typedef __attribute__((ext_vector_type(8)))  _Float16 v8h;
typedef __attribute__((ext_vector_type(8)))  float    v8f;

__device__ __forceinline__ float wave_sum(float v) {
#pragma unroll
  for (int off = 16; off > 0; off >>= 1) v += __shfl_xor(v, off, 32);
  return v;
}

// ---------------- f32 -> f16 conversions ----------------
__global__ __launch_bounds__(256) void cvt_x_f16(const float* __restrict__ in,
                                                 _Float16* __restrict__ out, int n) {
  int i = blockIdx.x * 256 + threadIdx.x;
  if (i < n) out[i] = (_Float16)in[i];
}

// Wg1h[i*D + d] = (f16) Wg[i*2D + d]   (first-half columns of Wg)
__global__ __launch_bounds__(256) void cvt_wg1_f16(const float* __restrict__ Wg,
                                                   _Float16* __restrict__ Wgh) {
  int e = blockIdx.x * 256 + threadIdx.x;       // 0 .. D*D-1
  int i = e >> 11, d = e & (DD - 1);
  Wgh[e] = (_Float16)Wg[(size_t)i * (2 * DD) + d];
}

// ---------------- cosine similarities (wave per row) ----------------
__global__ __launch_bounds__(256) void sims_kernel(const float* __restrict__ bank,
                                                   const float* __restrict__ ctx,
                                                   float* __restrict__ sims) {
  __shared__ float sctx[DD];
  for (int d = threadIdx.x; d < DD; d += 256) sctx[d] = ctx[d];
  __syncthreads();
  int wave = blockIdx.x * 8 + (threadIdx.x >> 5);
  int lane = threadIdx.x & 31;
  if (wave >= MM) return;
  const float* row = bank + (size_t)wave * DD;
  float dot = 0.f, nrm = 0.f;
  for (int d = lane; d < DD; d += 32) {
    float x = row[d];
    dot += x * sctx[d];
    nrm += x * x;
  }
  dot = wave_sum(dot);
  nrm = wave_sum(nrm);
  if (lane == 0) sims[wave] = dot / fmaxf(sqrtf(nrm), 1e-12f);
}

// ---------------- single-block top-5 + fold ctx norm & 1/sqrt(D) ----------------
__global__ __launch_bounds__(256) void topk_kernel(const float* __restrict__ sims,
                                                   const float* __restrict__ ctx,
                                                   int* __restrict__ top_idx,
                                                   float* __restrict__ sc) {
  __shared__ float rv[256];
  __shared__ int   ri[256];
  __shared__ int   sidx[KTOP];
  __shared__ float sval[KTOP];
  int tid = threadIdx.x;
  for (int p = 0; p < KTOP; ++p) {
    float best = -INFINITY; int bi = 0x7fffffff;
    for (int m = tid; m < MM; m += 256) {
      bool taken = false;
      for (int q = 0; q < p; ++q) taken |= (sidx[q] == m);
      if (taken) continue;
      float v = sims[m];
      if (v > best || (v == best && m < bi)) { best = v; bi = m; }
    }
    rv[tid] = best; ri[tid] = bi;
    __syncthreads();
    for (int s = 128; s > 0; s >>= 1) {
      if (tid < s) {
        if (rv[tid + s] > rv[tid] ||
            (rv[tid + s] == rv[tid] && ri[tid + s] < ri[tid])) {
          rv[tid] = rv[tid + s]; ri[tid] = ri[tid + s];
        }
      }
      __syncthreads();
    }
    if (tid == 0) { sidx[p] = ri[0]; sval[p] = rv[0]; }
    __syncthreads();
  }
  // context norm
  float s2 = 0.f;
  for (int d = tid; d < DD; d += 256) { float x = ctx[d]; s2 += x * x; }
  rv[tid] = s2;
  __syncthreads();
  for (int s = 128; s > 0; s >>= 1) { if (tid < s) rv[tid] += rv[tid + s]; __syncthreads(); }
  if (tid < KTOP) {
    float cn = fmaxf(sqrtf(rv[0]), 1e-12f);
    top_idx[tid] = sidx[tid];
    // fold cosine's 1/||ctx|| and the 1/sqrt(D) logit scale into one factor
    sc[tid] = (sval[tid] / cn) * 0.022097086912079608f; // 1/sqrt(2048)
  }
}

// ---------------- k = mem@Wk^T+bk, v = mem@Wv^T+bv (wave per output) ----------------
__global__ __launch_bounds__(256) void kv_kernel(const float* __restrict__ mvals,
                                                 const int* __restrict__ top_idx,
                                                 const float* __restrict__ Wk,
                                                 const float* __restrict__ bk,
                                                 const float* __restrict__ Wv,
                                                 const float* __restrict__ bv,
                                                 float* __restrict__ kmat,
                                                 float* __restrict__ vmat) {
  int gw = blockIdx.x * 8 + (threadIdx.x >> 5);  // 0 .. 2*K*D-1
  int lane = threadIdx.x & 31;
  int which = gw / (KTOP * DD);
  int rem = gw % (KTOP * DD);
  int j = rem / DD, i = rem % DD;
  const float* mrow = mvals + (size_t)top_idx[j] * DD;
  const float* wrow = (which ? Wv : Wk) + (size_t)i * DD;
  float acc = 0.f;
  for (int d = lane; d < DD; d += 32) acc += mrow[d] * wrow[d];
  acc = wave_sum(acc);
  if (lane == 0) {
    if (which) vmat[rem] = acc + bv[i];
    else       kmat[rem] = acc + bk[i];
  }
}

// ---------------- kproj[j,:] = k[j] @ Wq ; bqk[j] = bq . k[j]  (block per j) ----------------
__global__ __launch_bounds__(512) void kproj_kernel(const float* __restrict__ kmat,
                                                    const float* __restrict__ Wq,
                                                    const float* __restrict__ bq,
                                                    float* __restrict__ kproj,
                                                    float* __restrict__ bqk) {
  int j = blockIdx.x, tid = threadIdx.x;
  const float* krow = kmat + j * DD;
  float acc[4] = {0.f, 0.f, 0.f, 0.f};
  for (int i = 0; i < DD; ++i) {
    float kji = krow[i];
    const float* wrow = Wq + (size_t)i * DD;
#pragma unroll
    for (int t = 0; t < 4; ++t) acc[t] += kji * wrow[tid + t * 512];
  }
#pragma unroll
  for (int t = 0; t < 4; ++t) kproj[j * DD + tid + t * 512] = acc[t];

  __shared__ float red[512];
  float b = 0.f;
  for (int i = tid; i < DD; i += 512) b += bq[i] * krow[i];
  red[tid] = b;
  __syncthreads();
  for (int s = 256; s > 0; s >>= 1) { if (tid < s) red[tid] += red[tid + s]; __syncthreads(); }
  if (tid == 0) bqk[j] = red[0];
}

// ---------------- Vg[j,i] = v[j] . Wg[i, D:2D] (wave per output) ----------------
__global__ __launch_bounds__(256) void vg_kernel(const float* __restrict__ vmat,
                                                 const float* __restrict__ Wg,
                                                 float* __restrict__ Vg) {
  int gw = blockIdx.x * 8 + (threadIdx.x >> 5);  // 0 .. K*D-1
  int lane = threadIdx.x & 31;
  int j = gw / DD, i = gw % DD;
  const float* vrow = vmat + j * DD;
  const float* wrow = Wg + (size_t)i * (2 * DD) + DD;
  float acc = 0.f;
  for (int d = lane; d < DD; d += 32) acc += vrow[d] * wrow[d];
  acc = wave_sum(acc);
  if (lane == 0) Vg[gw] = acc;
}

// ---------------- scores + softmax (wave per batch row) ----------------
__global__ __launch_bounds__(256) void attn_kernel(const float* __restrict__ X,
                                                   const float* __restrict__ kproj,
                                                   const float* __restrict__ bqk,
                                                   const float* __restrict__ sc,
                                                   float* __restrict__ attn) {
  int b = blockIdx.x * 8 + (threadIdx.x >> 5);
  int lane = threadIdx.x & 31;
  const float* xr = X + (size_t)b * DD;
  float a0 = 0.f, a1 = 0.f, a2 = 0.f, a3 = 0.f, a4 = 0.f;
  for (int d = lane; d < DD; d += 32) {
    float x = xr[d];
    a0 += x * kproj[d];
    a1 += x * kproj[DD + d];
    a2 += x * kproj[2 * DD + d];
    a3 += x * kproj[3 * DD + d];
    a4 += x * kproj[4 * DD + d];
  }
  a0 = wave_sum(a0); a1 = wave_sum(a1); a2 = wave_sum(a2);
  a3 = wave_sum(a3); a4 = wave_sum(a4);
  if (lane == 0) {
    float s0 = (a0 + bqk[0]) * sc[0];
    float s1 = (a1 + bqk[1]) * sc[1];
    float s2 = (a2 + bqk[2]) * sc[2];
    float s3 = (a3 + bqk[3]) * sc[3];
    float s4 = (a4 + bqk[4]) * sc[4];
    float mx = fmaxf(fmaxf(fmaxf(s0, s1), fmaxf(s2, s3)), s4);
    float e0 = __expf(s0 - mx), e1 = __expf(s1 - mx), e2 = __expf(s2 - mx);
    float e3 = __expf(s3 - mx), e4 = __expf(s4 - mx);
    float inv = 1.f / (e0 + e1 + e2 + e3 + e4);
    float* ar = attn + b * KTOP;
    ar[0] = e0 * inv; ar[1] = e1 * inv; ar[2] = e2 * inv;
    ar[3] = e3 * inv; ar[4] = e4 * inv;
  }
}

// ---------------- WMMA gate GEMM with LDS staging + fully fused epilogue ----------
// Block tile 128x128 (8 waves = 4m x 2n, each wave 32x64). Per K-step (K=32) the
// block stages A(128x32) and B(128x32) f16 tiles in LDS (20KB of the 320KB WGP LDS),
// cutting L2 fragment traffic ~3x vs direct global fragment loads.
// Epilogue: gate = sigmoid(pre + attn@Vg + bg); attended = attn@v; out = X + gate*attended
__global__ __launch_bounds__(256) void gate_wmma_kernel(const _Float16* __restrict__ Xh,
                                                        const _Float16* __restrict__ Wgh,
                                                        const float* __restrict__ Xf,
                                                        const float* __restrict__ attn,
                                                        const float* __restrict__ Vg,
                                                        const float* __restrict__ vmat,
                                                        const float* __restrict__ bg,
                                                        float* __restrict__ out) {
  __shared__ _Float16 As[128 * LPITCH];
  __shared__ _Float16 Bs[128 * LPITCH];

  const int t    = threadIdx.x;
  const int lane = t & 31;
  const int wave = t >> 5;
  const int wm   = wave & 3;          // m sub-tile of this wave (0..3)
  const int wn   = wave >> 2;         // n sub-tile of this wave (0..1)
  const int m_blk = blockIdx.y * 128;
  const int n_blk = blockIdx.x * 128;

  const int lr    = lane & 15;               // row within 16-tile (ISA A/B layout)
  const int kbase = (lane < 16) ? 0 : 8;     // K sub-block per ISA 16-bit layout
  const int hi16  = lane >> 4;               // C/D layout: M = r + 8*hi16

  // staging indices: 512 (row,seg) slots, 2 per thread
  const int row0 = t >> 1;                   // 0..127
  const int seg0 = (t & 1) * 8;              // halves 0 / 8
  const int seg1 = seg0 + 16;                // halves 16 / 24

  v8f acc[2][4] = {};
  for (int k0 = 0; k0 < DD; k0 += 32) {
    // ---- cooperative stage of A/B K-slabs into LDS ----
    const size_t arow = (size_t)(m_blk + row0) * DD + k0;
    const size_t brow = (size_t)(n_blk + row0) * DD + k0;
    *(v8h*)&As[row0 * LPITCH + seg0] = *(const v8h*)(Xh + arow + seg0);
    *(v8h*)&As[row0 * LPITCH + seg1] = *(const v8h*)(Xh + arow + seg1);
    *(v8h*)&Bs[row0 * LPITCH + seg0] = *(const v8h*)(Wgh + brow + seg0);
    *(v8h*)&Bs[row0 * LPITCH + seg1] = *(const v8h*)(Wgh + brow + seg1);
    if (k0 + 32 < DD) {                      // global_prefetch_b8 of next K-slab
      __builtin_prefetch(Xh + arow + 32 + seg0, 0, 1);
      __builtin_prefetch(Wgh + brow + 32 + seg0, 0, 1);
    }
    __syncthreads();

    // ---- fragment loads from LDS + WMMA ----
    v16h a[2], b[4];
#pragma unroll
    for (int am = 0; am < 2; ++am) {
      const _Float16* p = &As[(wm * 32 + am * 16 + lr) * LPITCH + kbase];
      v8h lo = *(const v8h*)p;               // K = kbase .. kbase+7
      v8h hi = *(const v8h*)(p + 16);        // K = kbase+16 .. kbase+23
      a[am] = __builtin_shufflevector(lo, hi, 0, 1, 2, 3, 4, 5, 6, 7,
                                      8, 9, 10, 11, 12, 13, 14, 15);
    }
#pragma unroll
    for (int bn = 0; bn < 4; ++bn) {
      const _Float16* p = &Bs[(wn * 64 + bn * 16 + lr) * LPITCH + kbase];
      v8h lo = *(const v8h*)p;
      v8h hi = *(const v8h*)(p + 16);
      b[bn] = __builtin_shufflevector(lo, hi, 0, 1, 2, 3, 4, 5, 6, 7,
                                      8, 9, 10, 11, 12, 13, 14, 15);
    }
#pragma unroll
    for (int am = 0; am < 2; ++am)
#pragma unroll
      for (int bn = 0; bn < 4; ++bn)
        acc[am][bn] = __builtin_amdgcn_wmma_f32_16x16x32_f16(
            false, a[am], false, b[bn], (short)0, acc[am][bn], false, false);
    __syncthreads();
  }

#pragma unroll
  for (int bn = 0; bn < 4; ++bn) {
    const int n = n_blk + wn * 64 + bn * 16 + lr;
    const float bgn = bg[n];
    const float vg0 = Vg[n],          vg1 = Vg[DD + n],     vg2 = Vg[2 * DD + n];
    const float vg3 = Vg[3 * DD + n], vg4 = Vg[4 * DD + n];
    const float vv0 = vmat[n],          vv1 = vmat[DD + n],     vv2 = vmat[2 * DD + n];
    const float vv3 = vmat[3 * DD + n], vv4 = vmat[4 * DD + n];
#pragma unroll
    for (int am = 0; am < 2; ++am) {
#pragma unroll
      for (int r = 0; r < 8; ++r) {
        const int m = m_blk + wm * 32 + am * 16 + hi16 * 8 + r;
        const float* ar = attn + m * KTOP;
        const float a0 = ar[0], a1 = ar[1], a2 = ar[2], a3 = ar[3], a4 = ar[4];
        const float g = acc[am][bn][r] + bgn +
                        a0 * vg0 + a1 * vg1 + a2 * vg2 + a3 * vg3 + a4 * vg4;
        const float att = a0 * vv0 + a1 * vv1 + a2 * vv2 + a3 * vv3 + a4 * vv4;
        const float gate = 1.0f / (1.0f + __expf(-g));
        const size_t o = (size_t)m * DD + n;
        out[o] = Xf[o] + gate * att;
      }
    }
  }
}

extern "C" void kernel_launch(void* const* d_in, const int* in_sizes, int n_in,
                              void* d_out, int out_size, void* d_ws, size_t ws_size,
                              hipStream_t stream) {
  (void)in_sizes; (void)n_in; (void)out_size; (void)ws_size;
  const float* X    = (const float*)d_in[0];
  const float* ctx  = (const float*)d_in[1];
  const float* bank = (const float*)d_in[2];
  const float* mval = (const float*)d_in[3];
  const float* Wq   = (const float*)d_in[4];
  const float* bq   = (const float*)d_in[5];
  const float* Wk   = (const float*)d_in[6];
  const float* bk   = (const float*)d_in[7];
  const float* Wv   = (const float*)d_in[8];
  const float* bv   = (const float*)d_in[9];
  const float* Wg   = (const float*)d_in[10];
  const float* bg   = (const float*)d_in[11];
  float* out = (float*)d_out;

  // ---- workspace carve-up (256B aligned) ----
  char* base = (char*)d_ws;
  size_t off = 0;
  auto carve = [&](size_t bytes) {
    size_t o = off;
    off = (off + bytes + 255) & ~(size_t)255;
    return o;
  };
  float*    sims   = (float*)(base + carve(sizeof(float) * MM));
  int*      tidx   = (int*)  (base + carve(sizeof(int) * KTOP));
  float*    sc     = (float*)(base + carve(sizeof(float) * KTOP));
  float*    bqk    = (float*)(base + carve(sizeof(float) * KTOP));
  float*    kmat   = (float*)(base + carve(sizeof(float) * KTOP * DD));
  float*    vmat   = (float*)(base + carve(sizeof(float) * KTOP * DD));
  float*    kproj  = (float*)(base + carve(sizeof(float) * KTOP * DD));
  float*    Vg     = (float*)(base + carve(sizeof(float) * KTOP * DD));
  float*    attn   = (float*)(base + carve(sizeof(float) * BATCH * KTOP));
  _Float16* Xh     = (_Float16*)(base + carve(sizeof(_Float16) * (size_t)BATCH * DD));
  _Float16* Wgh    = (_Float16*)(base + carve(sizeof(_Float16) * (size_t)DD * DD));

  // 1) precision-convert the big-GEMM operands
  cvt_x_f16<<<(BATCH * DD) / 256, 256, 0, stream>>>(X, Xh, BATCH * DD);
  cvt_wg1_f16<<<(DD * DD) / 256, 256, 0, stream>>>(Wg, Wgh);

  // 2) cosine similarity scan (dominant HBM cost: 400 MB)
  sims_kernel<<<(MM + 7) / 8, 256, 0, stream>>>(bank, ctx, sims);

  // 3) top-5 selection + fold ctx-norm and 1/sqrt(D)
  topk_kernel<<<1, 256, 0, stream>>>(sims, ctx, tidx, sc);

  // 4) k, v projections of the 5 retrieved memories
  kv_kernel<<<(2 * KTOP * DD) / 8, 256, 0, stream>>>(mval, tidx, Wk, bk, Wv, bv, kmat, vmat);

  // 5) rank-5 algebraic fusions: kproj = k@Wq (+ bq.k), Vg = v@Wg2^T
  kproj_kernel<<<KTOP, 512, 0, stream>>>(kmat, Wq, bq, kproj, bqk);
  vg_kernel<<<(KTOP * DD) / 8, 256, 0, stream>>>(vmat, Wg, Vg);

  // 6) scores + softmax per batch row ([B,D]@[D,5], memory-bound)
  attn_kernel<<<BATCH / 8, 256, 0, stream>>>(X, kproj, bqk, sc, attn);

  // 7) the one remaining big GEMM (X@Wg1^T) on WMMA, LDS-staged, fused epilogue
  gate_wmma_kernel<<<dim3(DD / 128, BATCH / 128), 256, 0, stream>>>(
      Xh, Wgh, X, attn, Vg, vmat, bg, out);
}
